// MultiHeadAttention_10720238371032
// MI455X (gfx1250) — matrix operand order repaired and checked by reading, not compiled, not run on hardware
//
#include <hip/hip_runtime.h>
#include <math.h>

#define NH_HEADS 16
#define SEQ 1024
#define DM 1024
#define DQ 64
#define MROWS 8192   // 8 * 1024 rows

typedef __attribute__((ext_vector_type(16))) __bf16 bf16x16;
typedef __attribute__((ext_vector_type(8)))  float  f32x8;
typedef __attribute__((ext_vector_type(4))) unsigned int u32x4;
typedef __attribute__((ext_vector_type(8))) int i32x8;
typedef __attribute__((ext_vector_type(4))) int i32x4;

union FragBF { unsigned int u[8]; bf16x16 v; };
union Chunk16 { i32x4 v; unsigned short s[8]; };

// ---- CDNA5 feature detection (fallback keeps the build green) -------------
#if defined(__has_builtin)
# if __has_builtin(__builtin_amdgcn_global_load_async_to_lds_b128)
#  define HAVE_ASYNC_LDS 1
# endif
# if __has_builtin(__builtin_amdgcn_s_wait_asynccnt)
#  define HAVE_WAIT_ASYNC_BUILTIN 1
# endif
# if __has_builtin(__builtin_amdgcn_tensor_load_to_lds)
#  if (__clang_major__ >= 23)
#   define HAVE_TDM6 1   // amdgpu-toolchain (therock) 6-arg form
#  else
#   define HAVE_TDM5 1   // ROCm 7.2 clang-22 5-arg form
#  endif
# endif
#endif
#if HAVE_TDM5 || HAVE_TDM6
# define HAVE_TDM 1
#endif

// async global->LDS b128: typed addrspace pointers
#define GV4(p) ((__attribute__((address_space(1))) i32x4*)(p))
#define SV4(p) ((__attribute__((address_space(3))) i32x4*)(p))

#if HAVE_ASYNC_LDS
__device__ __forceinline__ void wait_async0() {
# if HAVE_WAIT_ASYNC_BUILTIN
  __builtin_amdgcn_s_wait_asynccnt(0);
# else
  asm volatile("s_wait_asynccnt 0" ::: "memory");
# endif
}
#endif

#if HAVE_TDM
__device__ __forceinline__ unsigned lds_off(void* p) {
  return (unsigned)(unsigned long long)(__attribute__((address_space(3))) char*)p;
}
#endif

__device__ __forceinline__ unsigned short f2bf(float f) {
  unsigned int u = __float_as_uint(f);
  u += 0x7FFFu + ((u >> 16) & 1u);           // round-to-nearest-even
  return (unsigned short)(u >> 16);
}

__device__ __forceinline__ f32x8 fzero8() {
  f32x8 z;
  #pragma unroll
  for (int e = 0; e < 8; ++e) z[e] = 0.0f;
  return z;
}

// ---------------------------------------------------------------------------
// Weight prep: Wq group-fold + fp32->bf16 for all weights; bq group-fold.
// ---------------------------------------------------------------------------
__global__ __launch_bounds__(256) void prep_kernel(
    const float* __restrict__ Wq, const float* __restrict__ bq,
    const float* __restrict__ Wk, const float* __restrict__ Wv,
    const float* __restrict__ Wo,
    unsigned short* __restrict__ wqe, unsigned short* __restrict__ wkb,
    unsigned short* __restrict__ wvb, unsigned short* __restrict__ wob,
    float* __restrict__ bqe) {
  int i = blockIdx.x * 256 + threadIdx.x;    // [0, 1024*1024)
  int d = i >> 10, c = i & 1023;
  wqe[i] = f2bf(Wq[(size_t)d * 2048 + c] + Wq[(size_t)d * 2048 + 1024 + c]);
  wkb[i] = f2bf(Wk[i]);
  wvb[i] = f2bf(Wv[i]);
  wob[i] = f2bf(Wo[i]);
  if (i < 1024) bqe[i] = bq[i] + bq[1024 + i];
}

// ---------------------------------------------------------------------------
// LayerNorm of q/k/v inputs -> bf16. One block per row; which input = b>>13.
// ---------------------------------------------------------------------------
__global__ __launch_bounds__(256) void ln_kernel(
    const float* __restrict__ q, const float* __restrict__ k,
    const float* __restrict__ v, const float* __restrict__ gamma,
    const float* __restrict__ beta, unsigned short* __restrict__ xq,
    unsigned short* __restrict__ xk, unsigned short* __restrict__ xv) {
  __shared__ float r1[256], r2[256];
  int b = blockIdx.x;
  int which = b >> 13;
  int row = b & 8191;
  const float* src = which == 0 ? q : (which == 1 ? k : v);
  unsigned short* dst = which == 0 ? xq : (which == 1 ? xk : xv);
  const float* x = src + (size_t)row * DM;
  int tid = threadIdx.x;
  float s = 0.f, ss = 0.f, vals[4];
  #pragma unroll
  for (int i = 0; i < 4; ++i) {
    float f = x[tid + i * 256];
    vals[i] = f; s += f; ss += f * f;
  }
  r1[tid] = s; r2[tid] = ss;
  __syncthreads();
  for (int off = 128; off > 0; off >>= 1) {
    if (tid < off) { r1[tid] += r1[tid + off]; r2[tid] += r2[tid + off]; }
    __syncthreads();
  }
  float mean = r1[0] * (1.0f / 1024.0f);
  float var  = r2[0] * (1.0f / 1024.0f) - mean * mean;
  float rstd = rsqrtf(var + 1e-5f);
  unsigned short* drow = dst + (size_t)row * DM;
  #pragma unroll
  for (int i = 0; i < 4; ++i) {
    int c = tid + i * 256;
    drow[c] = f2bf((vals[i] - mean) * rstd * gamma[c] + beta[c]);
  }
}

// ---------------------------------------------------------------------------
// Generic WMMA bf16 GEMM: C[8192,1024] = A[8192,1024] @ B[1024,1024] + bias.
// Tile 128x64, K-step 32. 8 waves, each owns a 16-row strip (4 WMMA n-tiles).
// A tile staged via async global->LDS; B^T staged via b128 loads + LDS scatter.
// mode 0: RoPE epilogue + head-major bf16 store (Q, K projections)
// mode 1: head-major bf16 store (V projection)
// mode 2: flat fp32 store (final output projection)
// ---------------------------------------------------------------------------
__global__ __launch_bounds__(256) void gemm128x64_bf16(
    const unsigned short* __restrict__ A, const unsigned short* __restrict__ B,
    const float* __restrict__ bias, void* __restrict__ outp, int mode) {
  __shared__ unsigned int   As[128 * 16];    // A tile [row][k-pair], 8 KB
  __shared__ unsigned short Bs[64 * 32];     // B^T tile [n][k], 4 KB
  __shared__ float          Cs[128 * 64];    // 32 KB

  const int tid  = threadIdx.x;
  const int lane = tid & 31;
  const int w    = tid >> 5;
  const int m    = lane & 15;
  const int half = lane >> 4;
  const int bm   = blockIdx.x;   // 64 row-tiles
  const int bn   = blockIdx.y;   // 16 col-tiles (== head index for modes 0/1)

  const unsigned int* Ag   = (const unsigned int*)A;
  unsigned int*       Bs_u = (unsigned int*)Bs;

  f32x8 acc[4];
  #pragma unroll
  for (int nc = 0; nc < 4; ++nc) acc[nc] = fzero8();

  for (int kt = 0; kt < DM; kt += 32) {
#if HAVE_ASYNC_LDS
    #pragma unroll
    for (int it = 0; it < 2; ++it) {         // A: 512 x 16B async chunks
      int e = tid + it * 256;
      int r = e >> 2, cu = e & 3;
      const char* g = (const char*)A + ((size_t)(bm * 128 + r) * DM + kt) * 2 + cu * 16;
      __builtin_amdgcn_global_load_async_to_lds_b128(
          GV4(g), SV4((char*)As + r * 64 + cu * 16), 0, 0);
    }
#else
    #pragma unroll
    for (int it = 0; it < 8; ++it) {         // stage A: 2048 dwords
      int e = tid + it * 256;
      int r = e >> 4, cu = e & 15;
      As[r * 16 + cu] = Ag[(size_t)(bm * 128 + r) * 512 + (kt >> 1) + cu];
    }
#endif
    {                                        // stage B^T: 256 x b128 chunks
      Chunk16 tb;
      int kk = tid >> 3, nn0 = (tid & 7) * 8;
      tb.v = *(const i32x4*)(B + (size_t)(kt + kk) * DM + bn * 64 + nn0);
      #pragma unroll
      for (int qq = 0; qq < 8; ++qq)
        Bs[(nn0 + qq) * 32 + kk] = tb.s[qq];
    }
    if (kt + 32 < DM) {                      // prefetch next K-step tiles
      __builtin_prefetch((const char*)A + ((size_t)(bm * 128 + (tid >> 1)) * DM + kt + 32) * 2, 0, 1);
      __builtin_prefetch((const char*)B + ((size_t)(kt + 32 + (tid >> 3)) * DM + bn * 64) * 2, 0, 1);
    }
#if HAVE_ASYNC_LDS
    wait_async0();
#endif
    __syncthreads();

    FragBF fa;                                // A frag: 16x32, rows w*16..+15
    #pragma unroll
    for (int v = 0; v < 8; ++v)
      fa.u[v] = As[(w * 16 + m) * 16 + (half << 3) + v];

    #pragma unroll
    for (int nc = 0; nc < 4; ++nc) {
      FragBF fb;                              // B frag: 32x16 at n0 = nc*16
      #pragma unroll
      for (int v = 0; v < 8; ++v)
        fb.u[v] = Bs_u[(nc * 16 + m) * 16 + (half << 3) + v];
      acc[nc] = __builtin_amdgcn_wmma_f32_16x16x32_bf16(
          false, fa.v, false, fb.v, (short)0, acc[nc], false, false);
    }
    __syncthreads();
  }

  #pragma unroll
  for (int nc = 0; nc < 4; ++nc)
    #pragma unroll
    for (int v = 0; v < 8; ++v)
      Cs[(w * 16 + v + half * 8) * 64 + nc * 16 + m] = acc[nc][v];
  __syncthreads();

  for (int it = 0; it < 32; ++it) {
    int e = tid + it * 256;                  // [0, 8192)
    int r = e >> 6, c = e & 63;
    int Rg = bm * 128 + r;
    int Cg = bn * 64 + c;
    float x = Cs[r * 64 + c] + bias[Cg];
    if (mode == 2) {
      ((float*)outp)[(size_t)Rg * DM + Cg] = x;
    } else {
      int n = Rg >> 10, s = Rg & 1023;
      float y = x;
      if (mode == 0) {                       // interleaved RoPE, head dim == 64
        float xp = Cs[r * 64 + (c ^ 1)] + bias[Cg ^ 1];
        int p = c >> 1;
        float inv = __expf(-(float)(2 * p) * (9.210340371976184f / 64.0f));
        float sn, cs;
        __sincosf((float)s * inv, &sn, &cs);
        y = (c & 1) ? (x * cs + xp * sn) : (x * cs - xp * sn);
      }
      // head-major: [n][h][s][d], h == bn, d == c
      ((unsigned short*)outp)[(((size_t)n * NH_HEADS + bn) * SEQ + s) * DQ + c] =
          f2bf(y);
    }
  }
}

// ---------------------------------------------------------------------------
// Flash attention: block = (n,h) x 128 query rows; 8 waves x 16-row strips.
// K tile staged by the Tensor Data Mover (TDM) when available (else async
// global->LDS, else sync). V^T staged via b128 loads + LDS scatter.
// Softcap tanh(30), causal + padding mask, running softmax in C-frag layout.
// ---------------------------------------------------------------------------
__global__ __launch_bounds__(256) void attn_kernel(
    const unsigned short* __restrict__ Qb, const unsigned short* __restrict__ Kb,
    const unsigned short* __restrict__ Vb, const unsigned char* __restrict__ kpm,
    unsigned short* __restrict__ AO) {
  __shared__ unsigned short Kt[32 * 64];     // K tile [key][dq], 4 KB
  __shared__ unsigned short Vt[64 * 32];     // V^T tile [dq][key], 4 KB
  __shared__ unsigned short Pl[8 * 16 * 32]; // per-wave P staging, 8 KB

  const int tid = threadIdx.x, lane = tid & 31, w = tid >> 5;
  const int m = lane & 15, half = lane >> 4;
  const int n = blockIdx.x >> 4, h = blockIdx.x & 15;
  const int q0 = blockIdx.y * 128;
  const size_t hb = ((size_t)n * NH_HEADS + h) * SEQ;

  const unsigned int* Qu   = (const unsigned int*)Qb + (hb + q0 + w * 16) * 32;
  unsigned int*       Kt_u = (unsigned int*)Kt;
  unsigned int*       Vt_u = (unsigned int*)Vt;
  unsigned int*       Pl_u = (unsigned int*)Pl;
#if !(HAVE_TDM) && !(HAVE_ASYNC_LDS)
  const unsigned int* Ku   = (const unsigned int*)Kb;
#endif

  FragBF qf[2];                               // Q rows in A-fragment layout
  #pragma unroll
  for (int kc = 0; kc < 2; ++kc)
    #pragma unroll
    for (int v = 0; v < 8; ++v)
      qf[kc].u[v] =
          Qu[(size_t)m * 32 + kc * 16 + (half << 3) + v];

  f32x8 oacc[4];
  #pragma unroll
  for (int nc = 0; nc < 4; ++nc) oacc[nc] = fzero8();
  float mi[8], li[8];
  #pragma unroll
  for (int v = 0; v < 8; ++v) { mi[v] = -1e30f; li[v] = 0.0f; }

  const int nj = (q0 + 128) >> 5;            // causal: cover rows up to q0+127
  const int myrow0 = q0 + w * 16;
  const unsigned char* kp = kpm + (size_t)n * SEQ;

  for (int j = 0; j < nj; ++j) {
    // ---- stage K tile [32][64] bf16 -------------------------------------
#if HAVE_TDM
    if (w == 0) {
      unsigned long long ga =
          (unsigned long long)(const char*)Kb + (hb + (size_t)j * 32) * (DQ * 2);
      u32x4 g0;
      g0[0] = 1u;                                          // count=1, user D#
      g0[1] = lds_off(Kt);                                 // LDS byte address
      g0[2] = (unsigned)(ga & 0xFFFFFFFFull);              // global addr lo
      g0[3] = (unsigned)((ga >> 32) & 0x1FFFFFFull) | (2u << 30); // hi | type=2
      i32x8 g1;
      g1[0] = 0x00010000;          // workgroup_mask=0, data_size=1 (2 bytes)
      g1[1] = (DQ & 0xFFFF) << 16; // tensor_dim0 = 64 (low 16 @ bits 63:48)
      g1[2] = (SEQ << 16);         // dim0 hi=0, tensor_dim1 = 1024 (bits 95:80)
      g1[3] = (DQ << 16);          // dim1 hi=0, tile_dim0 = 64 (bits 127:112)
      g1[4] = 32;                  // tile_dim1 = 32, tile_dim2 = 0
      g1[5] = DQ;                  // tensor_dim0_stride = 64
      g1[6] = 0;
      g1[7] = 0;
      i32x4 gz;
      gz[0] = 0; gz[1] = 0; gz[2] = 0; gz[3] = 0;
# if HAVE_TDM5
      __builtin_amdgcn_tensor_load_to_lds(g0, g1, gz, gz, 0);
# else
      i32x8 gz8;
      #pragma unroll
      for (int e = 0; e < 8; ++e) gz8[e] = 0;
      __builtin_amdgcn_tensor_load_to_lds(g0, g1, gz, gz, gz8, 0);
# endif
      __builtin_amdgcn_s_wait_tensorcnt(0);
    }
#elif HAVE_ASYNC_LDS
    {
      int r = tid >> 3, cu = tid & 7;        // 256 x 16B async chunks
      const char* g = (const char*)Kb + (hb + (size_t)j * 32 + r) * (DQ * 2) + cu * 16;
      __builtin_amdgcn_global_load_async_to_lds_b128(
          GV4(g), SV4((char*)Kt + r * 128 + cu * 16), 0, 0);
    }
#else
    #pragma unroll
    for (int it = 0; it < 4; ++it) {         // stage K tile: 1024 dwords
      int e = tid + it * 256;
      int r = e >> 5, cu = e & 31;
      Kt_u[r * 32 + cu] = Ku[(hb + j * 32 + r) * 32 + cu];
    }
#endif
    // ---- stage V^T tile [64][32] bf16: 256 x b128 chunks + LDS scatter --
    {
      Chunk16 tv;
      int key = tid >> 3, d0 = (tid & 7) * 8;
      tv.v = *(const i32x4*)(Vb + (hb + (size_t)j * 32 + key) * DQ + d0);
      #pragma unroll
      for (int qq = 0; qq < 8; ++qq)
        Vt[(d0 + qq) * 32 + key] = tv.s[qq];
    }
#if HAVE_ASYNC_LDS && !(HAVE_TDM)
    wait_async0();
#endif
    __syncthreads();

    // S = Q @ K^T : two 16x16 n-tiles, K-dim 64 in two 32-chunks
    f32x8 sa0 = fzero8(), sa1 = fzero8();
    #pragma unroll
    for (int kc = 0; kc < 2; ++kc) {
      FragBF fb;
      #pragma unroll
      for (int v = 0; v < 8; ++v)
        fb.u[v] = Kt_u[m * 32 + kc * 16 + (half << 3) + v];
      sa0 = __builtin_amdgcn_wmma_f32_16x16x32_bf16(
          false, qf[kc].v, false, fb.v, (short)0, sa0, false, false);
      #pragma unroll
      for (int v = 0; v < 8; ++v)
        fb.u[v] = Kt_u[(16 + m) * 32 + kc * 16 + (half << 3) + v];
      sa1 = __builtin_amdgcn_wmma_f32_16x16x32_bf16(
          false, qf[kc].v, false, fb.v, (short)0, sa1, false, false);
    }

    const int col0 = j * 32 + m;
    const int col1 = col0 + 16;
    const bool pad0 = kp[col0] != 0;
    const bool pad1 = kp[col1] != 0;
    #pragma unroll
    for (int v = 0; v < 8; ++v) {
      int row = myrow0 + v + half * 8;
      float a0 = sa0[v] * 0.125f;
      a0 = 30.0f * tanhf(a0 * (1.0f / 30.0f));
      if (col0 > row || pad0) a0 = -1e30f;
      float a1 = sa1[v] * 0.125f;
      a1 = 30.0f * tanhf(a1 * (1.0f / 30.0f));
      if (col1 > row || pad1) a1 = -1e30f;
      float mx = fmaxf(a0, a1);
      #pragma unroll
      for (int xm = 8; xm >= 1; xm >>= 1) mx = fmaxf(mx, __shfl_xor(mx, xm, 32));
      float mnew = fmaxf(mi[v], mx);
      float f  = __expf(mi[v] - mnew);
      float p0 = __expf(a0 - mnew);
      float p1 = __expf(a1 - mnew);
      float ps = p0 + p1;
      #pragma unroll
      for (int xm = 8; xm >= 1; xm >>= 1) ps += __shfl_xor(ps, xm, 32);
      li[v] = li[v] * f + ps;
      mi[v] = mnew;
      #pragma unroll
      for (int nc = 0; nc < 4; ++nc) oacc[nc][v] *= f;
      Pl[w * 512 + (v + half * 8) * 32 + m]      = f2bf(p0);
      Pl[w * 512 + (v + half * 8) * 32 + 16 + m] = f2bf(p1);
    }
    // wave-local producer->consumer through LDS: wait for DS stores
    asm volatile("s_wait_dscnt 0" ::: "memory");

    FragBF pf;                                // P as 16x32 A fragment
    #pragma unroll
    for (int v = 0; v < 8; ++v)
      pf.u[v] = Pl_u[w * 256 + m * 16 + (half << 3) + v];

    #pragma unroll
    for (int nc = 0; nc < 4; ++nc) {
      FragBF fb;
      #pragma unroll
      for (int v = 0; v < 8; ++v)
        fb.u[v] = Vt_u[(nc * 16 + m) * 16 + (half << 3) + v];
      oacc[nc] = __builtin_amdgcn_wmma_f32_16x16x32_bf16(
          false, pf.v, false, fb.v, (short)0, oacc[nc], false, false);
    }
    __syncthreads();
  }

  #pragma unroll
  for (int v = 0; v < 8; ++v) {
    float rinv = li[v] > 0.f ? 1.0f / li[v] : 0.f;
    int row = myrow0 + v + half * 8;
    #pragma unroll
    for (int nc = 0; nc < 4; ++nc) {
      int col = h * 64 + nc * 16 + m;
      AO[((size_t)n * SEQ + row) * DM + col] = f2bf(oacc[nc][v] * rinv);
    }
  }
}

// ---------------------------------------------------------------------------
extern "C" void kernel_launch(void* const* d_in, const int* in_sizes, int n_in,
                              void* d_out, int out_size, void* d_ws,
                              size_t ws_size, hipStream_t stream) {
  (void)in_sizes; (void)n_in; (void)out_size; (void)ws_size;
  const float* qs    = (const float*)d_in[0];
  const float* ks    = (const float*)d_in[1];
  const float* vs    = (const float*)d_in[2];
  const unsigned char* kpm = (const unsigned char*)d_in[3];
  const float* gamma = (const float*)d_in[4];
  const float* beta  = (const float*)d_in[5];
  const float* Wq    = (const float*)d_in[6];
  const float* bq    = (const float*)d_in[7];
  const float* Wk    = (const float*)d_in[8];
  const float* bk    = (const float*)d_in[9];
  const float* Wv    = (const float*)d_in[10];
  const float* bv    = (const float*)d_in[11];
  const float* Wo    = (const float*)d_in[12];
  const float* bo    = (const float*)d_in[13];

  char* ws = (char*)d_ws;
  const size_t MB = 1ull << 20;
  unsigned short* xlnq = (unsigned short*)(ws +   0 * MB);  // 16 MB
  unsigned short* xlnk = (unsigned short*)(ws +  16 * MB);  // 16 MB
  unsigned short* xlnv = (unsigned short*)(ws +  32 * MB);  // 16 MB
  unsigned short* wqe  = (unsigned short*)(ws +  48 * MB);  //  2 MB
  unsigned short* wkb  = (unsigned short*)(ws +  50 * MB);  //  2 MB
  unsigned short* wvb  = (unsigned short*)(ws +  52 * MB);  //  2 MB
  unsigned short* wob  = (unsigned short*)(ws +  54 * MB);  //  2 MB
  float*          bqe  = (float*)         (ws +  56 * MB);  //  4 KB
  unsigned short* Qb   = (unsigned short*)(ws +  57 * MB);  // 16 MB
  unsigned short* Kb   = (unsigned short*)(ws +  73 * MB);  // 16 MB
  unsigned short* Vb   = (unsigned short*)(ws +  89 * MB);  // 16 MB
  unsigned short* AO   = (unsigned short*)(ws + 105 * MB);  // 16 MB

  prep_kernel<<<4096, 256, 0, stream>>>(Wq, bq, Wk, Wv, Wo,
                                        wqe, wkb, wvb, wob, bqe);
  ln_kernel<<<3 * 8192, 256, 0, stream>>>(qs, ks, vs, gamma, beta,
                                          xlnq, xlnk, xlnv);
  gemm128x64_bf16<<<dim3(64, 16), 256, 0, stream>>>(xlnq, wqe, bqe, (void*)Qb, 0);
  gemm128x64_bf16<<<dim3(64, 16), 256, 0, stream>>>(xlnk, wkb, bk,  (void*)Kb, 0);
  gemm128x64_bf16<<<dim3(64, 16), 256, 0, stream>>>(xlnv, wvb, bv,  (void*)Vb, 1);
  attn_kernel<<<dim3(128, 8), 256, 0, stream>>>(Qb, Kb, Vb, kpm, AO);
  gemm128x64_bf16<<<dim3(64, 16), 256, 0, stream>>>(AO, wob, bo, d_out, 2);
}
